// MultiHeadAttention_21629455303350
// MI455X (gfx1250) — matrix operand order repaired and checked by reading, not compiled, not run on hardware
//
#include <hip/hip_runtime.h>

typedef __bf16 bf16;
typedef __attribute__((ext_vector_type(16))) __bf16 v16bf;
typedef __attribute__((ext_vector_type(8)))  __bf16 v8bf;
typedef __attribute__((ext_vector_type(8)))  float  v8f;
typedef __attribute__((ext_vector_type(4)))  unsigned int u32x4;
typedef __attribute__((ext_vector_type(4)))  int i32x4;
typedef __attribute__((ext_vector_type(8)))  int i32x8;

static __device__ __forceinline__ bf16 f2bf(float f) {
  unsigned u = __builtin_bit_cast(unsigned, f);
  unsigned r = u + 0x7FFFu + ((u >> 16) & 1u);   // round-to-nearest-even
  unsigned short h = (unsigned short)(r >> 16);
  return __builtin_bit_cast(bf16, h);
}

// A-fragment (16x32, bf16): lane L<16 -> row L, K {0..7, 16..23}; lane L+16 -> K {8..15, 24..31}
static __device__ __forceinline__ v16bf load_fragA(const bf16* base, int ld, int lane) {
  const int r = lane & 15, h = (lane >> 4) & 1;
  const bf16* p = base + (size_t)r * ld + h * 8;
  v8bf lo = *(const v8bf*)p;          // K  h*8 .. h*8+7
  v8bf hi = *(const v8bf*)(p + 16);   // K 16+h*8 .. 16+h*8+7
  v16bf out;
#pragma unroll
  for (int i = 0; i < 8; ++i) { out[i] = lo[i]; out[8 + i] = hi[i]; }
  return out;
}

// B-fragment (32x16, bf16), source stored transposed (Bt[n][k] row-major):
// lane L<16 -> col N=L, K {0..7, 16..23}; lane L+16 -> K {8..15, 24..31}
static __device__ __forceinline__ v16bf load_fragB(const bf16* baseT, int ld, int lane) {
  const int n = lane & 15, h = (lane >> 4) & 1;
  const bf16* p = baseT + (size_t)n * ld + h * 8;
  v8bf lo = *(const v8bf*)p;
  v8bf hi = *(const v8bf*)(p + 16);
  v16bf out;
#pragma unroll
  for (int i = 0; i < 8; ++i) { out[i] = lo[i]; out[8 + i] = hi[i]; }
  return out;
}

static __device__ __forceinline__ v8f wmma_bf16(v16bf a, v16bf b, v8f c) {
  return __builtin_amdgcn_wmma_f32_16x16x32_bf16(false, a, false, b, (short)0, c, false, false);
}

static __device__ __forceinline__ float hmax16(float v) {
  v = fmaxf(v, __shfl_xor(v, 8, 32));
  v = fmaxf(v, __shfl_xor(v, 4, 32));
  v = fmaxf(v, __shfl_xor(v, 2, 32));
  v = fmaxf(v, __shfl_xor(v, 1, 32));
  return v;
}
static __device__ __forceinline__ float hsum16(float v) {
  v += __shfl_xor(v, 8, 32);
  v += __shfl_xor(v, 4, 32);
  v += __shfl_xor(v, 2, 32);
  v += __shfl_xor(v, 1, 32);
  return v;
}

// ---- Tensor Data Mover: 2D tile (64 rows x 128B) global -> LDS, +16B/row pad -> 144B stride ----
// D# per CDNA5 ISA ch.8: group0 {count,lds_addr,global_addr,type=2},
// group1 {data_size=2B, pad 4dw every 32dw, tensor_dim0=64, dim1=1024, tile 64x64, dim0_stride=64}
static __device__ __forceinline__ void tdm_load_k64(unsigned lds_off, const bf16* gptr) {
  unsigned long long ga = (unsigned long long)(uintptr_t)gptr;
  u32x4 g0;
  g0[0] = 1u;                                                   // count=1 (valid user D#)
  g0[1] = lds_off;                                              // lds_addr (bytes)
  g0[2] = (unsigned)(ga & 0xFFFFFFFFu);                         // global_addr[31:0]
  g0[3] = (unsigned)((ga >> 32) & 0x01FFFFFFu) | (2u << 30);    // global_addr[56:32] | type=2
  i32x8 g1;
  g1[0] = (int)((1u << 16) | (1u << 20) | (4u << 22) | (3u << 25)); // data_size=2B, pad_en, ivl=32dw, amt=4dw
  g1[1] = (int)(64u << 16);                                     // tensor_dim0 = 64
  g1[2] = (int)(1024u << 16);                                   // tensor_dim1 = 1024
  g1[3] = (int)(64u << 16);                                     // tile_dim0 = 64
  g1[4] = 64;                                                   // tile_dim1 = 64, tile_dim2 = 0
  g1[5] = 64;                                                   // tensor_dim0_stride = 64
  g1[6] = 0;
  g1[7] = 0;
  i32x4 z4; z4[0] = z4[1] = z4[2] = z4[3] = 0;
#if __clang_major__ >= 23
  i32x8 z8;
#pragma unroll
  for (int i = 0; i < 8; ++i) z8[i] = 0;
  __builtin_amdgcn_tensor_load_to_lds(g0, g1, z4, z4, z8, 0);
#else
  __builtin_amdgcn_tensor_load_to_lds(g0, g1, z4, z4, 0);
#endif
}

// ---------------- f32 -> bf16 conversion ----------------
__global__ __launch_bounds__(256) void cvt_f32_bf16(const float* __restrict__ in,
                                                    bf16* __restrict__ out, int n) {
  for (int i = blockIdx.x * blockDim.x + threadIdx.x; i < n; i += gridDim.x * blockDim.x)
    out[i] = f2bf(in[i]);
}

// ---------------- bf16 GEMM: C[row,col] = sum_k A[row,k]*B[col,k] ----------------
// mode 0: Cout = float [M][N]
// mode 1: head-split bf16: row -> (b=row>>10, l=row&1023), col -> (head=col&15, d=col>>4)
__global__ __launch_bounds__(256) void gemm_bf16_nt(const bf16* __restrict__ A,
                                                    const bf16* __restrict__ B,
                                                    void* __restrict__ Cout,
                                                    int M, int N, int K, int mode, float scale) {
  constexpr int LDA = 72;  // 64-deep K stage + 8 pad
  __shared__ __align__(16) bf16 As[128 * LDA];
  __shared__ __align__(16) bf16 Bs[64 * LDA];
  const int tid = threadIdx.x, lane = tid & 31, wid = tid >> 5;
  const int mbase = blockIdx.y * 128, nbase = blockIdx.x * 64;

  v8f acc[4];
#pragma unroll
  for (int ng = 0; ng < 4; ++ng)
#pragma unroll
    for (int r = 0; r < 8; ++r) acc[ng][r] = 0.f;

  v8bf ra[4], rb[2];
  auto gload = [&](int kt) {
#pragma unroll
    for (int j = 0; j < 4; ++j) {            // 1024 chunks of 8 bf16 for As
      int c = tid + j * 256, row = c >> 3, ch = c & 7;
      ra[j] = *(const v8bf*)&A[(size_t)(mbase + row) * K + kt + ch * 8];
    }
#pragma unroll
    for (int j = 0; j < 2; ++j) {            // 512 chunks for Bs
      int c = tid + j * 256, row = c >> 3, ch = c & 7;
      rb[j] = *(const v8bf*)&B[(size_t)(nbase + row) * K + kt + ch * 8];
    }
  };
  auto lstore = [&]() {
#pragma unroll
    for (int j = 0; j < 4; ++j) {
      int c = tid + j * 256, row = c >> 3, ch = c & 7;
      *(v8bf*)&As[row * LDA + ch * 8] = ra[j];
    }
#pragma unroll
    for (int j = 0; j < 2; ++j) {
      int c = tid + j * 256, row = c >> 3, ch = c & 7;
      *(v8bf*)&Bs[row * LDA + ch * 8] = rb[j];
    }
  };

  gload(0);
  for (int kt = 0; kt < K; kt += 64) {
    lstore();
    __syncthreads();
    if (kt + 64 < K) gload(kt + 64);   // overlap next-stage global loads with compute
    v16bf a0 = load_fragA(&As[(wid * 16) * LDA], LDA, lane);
    v16bf a1 = load_fragA(&As[(wid * 16) * LDA + 32], LDA, lane);
#pragma unroll
    for (int ng = 0; ng < 4; ++ng) {
      v16bf b0 = load_fragB(&Bs[(ng * 16) * LDA], LDA, lane);
      v16bf b1 = load_fragB(&Bs[(ng * 16) * LDA + 32], LDA, lane);
      acc[ng] = wmma_bf16(a0, b0, acc[ng]);
      acc[ng] = wmma_bf16(a1, b1, acc[ng]);
    }
    __syncthreads();
  }

  const int nloc = lane & 15, hh = lane >> 4;
  if (mode == 0) {
    float* C = (float*)Cout;
#pragma unroll
    for (int ng = 0; ng < 4; ++ng)
#pragma unroll
      for (int r = 0; r < 8; ++r) {
        int row = mbase + wid * 16 + hh * 8 + r;
        int col = nbase + ng * 16 + nloc;
        C[(size_t)row * N + col] = acc[ng][r];
      }
  } else {
    bf16* C = (bf16*)Cout;
#pragma unroll
    for (int ng = 0; ng < 4; ++ng)
#pragma unroll
      for (int r = 0; r < 8; ++r) {
        int row = mbase + wid * 16 + hh * 8 + r;
        int col = nbase + ng * 16 + nloc;
        int bb = row >> 10, ll = row & 1023;
        int head = col & 15, d = col >> 4;
        C[(size_t)((bb * 16 + head) * 1024 + ll) * 64 + d] = f2bf(acc[ng][r] * scale);
      }
  }
}

// ---------------- flash attention (attends over K, per reference bug) ----------------
// Qh/Kh: [b*16 heads][1024 rows][64] bf16 (Q pre-scaled by 1/sqrt(64))
// Xout:  [b*1024][1024] bf16, col = d*16 + head
__global__ __launch_bounds__(256) void flash_attn_kk(const bf16* __restrict__ Qh,
                                                     const bf16* __restrict__ Kh,
                                                     bf16* __restrict__ Xout) {
  constexpr int LDK = 72;                          // 144B row stride == TDM pad output
  __shared__ __align__(16) bf16 Kr[64 * LDK];      // K chunk rows m, cols d (TDM-filled)
  __shared__ __align__(16) bf16 Kt[64 * LDK];      // transposed: rows d, cols m
  __shared__ __align__(16) bf16 Ps[8][16 * LDK];   // per-wave P tile 16 x 64

  const int tid = threadIdx.x, lane = tid & 31, wid = tid >> 5;
  const int nloc = lane & 15, hh = lane >> 4;
  const int bh = blockIdx.y;           // b*16 + head
  const int b = bh >> 4, head = bh & 15;
  const int q0 = blockIdx.x * 128 + wid * 16;

  const bf16* Qbase = Qh + (size_t)bh * 1024 * 64 + (size_t)q0 * 64;
  const bf16* Kbase = Kh + (size_t)bh * 1024 * 64;
  const unsigned kr_lds = (unsigned)(uintptr_t)&Kr[0];   // generic LDS ptr: low 32 bits = offset

  v16bf qa0 = load_fragA(Qbase, 64, lane);
  v16bf qa1 = load_fragA(Qbase + 32, 64, lane);

  float mi[8], li[8];
  v8f acc[4];
#pragma unroll
  for (int r = 0; r < 8; ++r) { mi[r] = -1e30f; li[r] = 0.f; }
#pragma unroll
  for (int dg = 0; dg < 4; ++dg)
#pragma unroll
    for (int r = 0; r < 8; ++r) acc[dg][r] = 0.f;

  for (int kb = 0; kb < 1024; kb += 64) {
    // ---- TDM: one wave DMAs the 64x64 K tile into Kr (padded rows) ----
    if (wid == 0) {
      tdm_load_k64(kr_lds, Kbase + (size_t)kb * 64);
      __builtin_amdgcn_s_wait_tensorcnt(0);
    }
    if (kb + 64 < 1024)
      __builtin_prefetch(&Kbase[(size_t)(kb + 64) * 64 + tid * 16], 0, 0);
    __syncthreads();

    // build transposed copy Kt[d][m] from Kr[m][d] (LDS->LDS)
#pragma unroll
    for (int c = tid; c < 512; c += 256) {
      int m = c >> 3, ch = c & 7;
      v8bf v = *(const v8bf*)&Kr[m * LDK + ch * 8];
#pragma unroll
      for (int i = 0; i < 8; ++i) Kt[(ch * 8 + i) * LDK + m] = v[i];
    }
    __syncthreads();

    // S = Q @ K^T for this chunk (4 fragments of 16 keys)
    v8f sfr[4];
    float newm[8];
#pragma unroll
    for (int r = 0; r < 8; ++r) newm[r] = mi[r];
#pragma unroll
    for (int ng = 0; ng < 4; ++ng) {
      v16bf b0 = load_fragB(&Kr[(ng * 16) * LDK], LDK, lane);
      v16bf b1 = load_fragB(&Kr[(ng * 16) * LDK + 32], LDK, lane);
      v8f s;
#pragma unroll
      for (int r = 0; r < 8; ++r) s[r] = 0.f;
      s = wmma_bf16(qa0, b0, s);
      s = wmma_bf16(qa1, b1, s);
      sfr[ng] = s;
#pragma unroll
      for (int r = 0; r < 8; ++r) newm[r] = fmaxf(newm[r], hmax16(s[r]));
    }

    // online softmax rescale
    float psum[8];
#pragma unroll
    for (int r = 0; r < 8; ++r) {
      float sc = __expf(mi[r] - newm[r]);
      li[r] *= sc;
#pragma unroll
      for (int dg = 0; dg < 4; ++dg) acc[dg][r] *= sc;
      mi[r] = newm[r];
      psum[r] = 0.f;
    }

    // P = exp(S - m), C-layout -> LDS -> A-layout
    bf16* Pw = Ps[wid];
#pragma unroll
    for (int ng = 0; ng < 4; ++ng)
#pragma unroll
      for (int r = 0; r < 8; ++r) {
        float p = __expf(sfr[ng][r] - mi[r]);
        psum[r] += p;
        Pw[(hh * 8 + r) * LDK + ng * 16 + nloc] = f2bf(p);
      }
#pragma unroll
    for (int r = 0; r < 8; ++r) li[r] += hsum16(psum[r]);

    v16bf pa0 = load_fragA(Pw, LDK, lane);
    v16bf pa1 = load_fragA(Pw + 32, LDK, lane);
#pragma unroll
    for (int dg = 0; dg < 4; ++dg) {
      v16bf kb0 = load_fragB(&Kt[(dg * 16) * LDK], LDK, lane);
      v16bf kb1 = load_fragB(&Kt[(dg * 16) * LDK + 32], LDK, lane);
      acc[dg] = wmma_bf16(pa0, kb0, acc[dg]);
      acc[dg] = wmma_bf16(pa1, kb1, acc[dg]);
    }
    __syncthreads();   // protect Kr/Kt before next TDM overwrite
  }

  // normalize and scatter to interleaved-hidden layout: col = d*16 + head
#pragma unroll
  for (int dg = 0; dg < 4; ++dg)
#pragma unroll
    for (int r = 0; r < 8; ++r) {
      int qg = q0 + hh * 8 + r;
      int d = dg * 16 + nloc;
      float val = acc[dg][r] / li[r];
      Xout[(size_t)(b * 1024 + qg) * 1024 + d * 16 + head] = f2bf(val);
    }
}

extern "C" void kernel_launch(void* const* d_in, const int* in_sizes, int n_in,
                              void* d_out, int out_size, void* d_ws, size_t ws_size,
                              hipStream_t stream) {
  (void)in_sizes; (void)n_in; (void)out_size; (void)ws_size;
  const float* q  = (const float*)d_in[0];
  const float* k  = (const float*)d_in[1];
  // d_in[2] (v) and d_in[5] (w_v) are dead: vp never affects the output.
  const float* wq = (const float*)d_in[3];
  const float* wk = (const float*)d_in[4];
  const float* wo = (const float*)d_in[6];

  char* ws = (char*)d_ws;
  const size_t MB = (size_t)1 << 20;
  bf16* qbf = (bf16*)(ws + 0 * MB);    // 8 MB  [4096][1024]
  bf16* kbf = (bf16*)(ws + 8 * MB);    // 8 MB
  bf16* wqb = (bf16*)(ws + 16 * MB);   // 2 MB
  bf16* wkb = (bf16*)(ws + 18 * MB);   // 2 MB
  bf16* wob = (bf16*)(ws + 20 * MB);   // 2 MB
  bf16* qh  = (bf16*)(ws + 22 * MB);   // 8 MB  [64 bh][1024][64]
  bf16* kh  = (bf16*)(ws + 30 * MB);   // 8 MB
  bf16* xbf = (bf16*)(ws + 38 * MB);   // 8 MB  [4096][1024]

  const int n4m = 4 * 1024 * 1024, n1m = 1024 * 1024;
  cvt_f32_bf16<<<512, 256, 0, stream>>>(q, qbf, n4m);
  cvt_f32_bf16<<<512, 256, 0, stream>>>(k, kbf, n4m);
  cvt_f32_bf16<<<256, 256, 0, stream>>>(wq, wqb, n1m);
  cvt_f32_bf16<<<256, 256, 0, stream>>>(wk, wkb, n1m);
  cvt_f32_bf16<<<256, 256, 0, stream>>>(wo, wob, n1m);

  dim3 g(16, 32), blk(256);
  gemm_bf16_nt<<<g, blk, 0, stream>>>(qbf, wqb, (void*)qh, 4096, 1024, 1024, 1, 0.125f);
  gemm_bf16_nt<<<g, blk, 0, stream>>>(kbf, wkb, (void*)kh, 4096, 1024, 1024, 1, 1.0f);
  flash_attn_kk<<<dim3(8, 64), blk, 0, stream>>>(qh, kh, xbf);
  gemm_bf16_nt<<<g, blk, 0, stream>>>(xbf, wob, d_out, 4096, 1024, 1024, 0, 1.0f);
}
